// HedgehogAttention_49787260895258
// MI455X (gfx1250) — matrix operand order repaired and checked by reading, not compiled, not run on hardware
//
#include <hip/hip_runtime.h>
#include <hip/hip_bf16.h>

typedef __attribute__((ext_vector_type(16))) _Float16 v16h;
typedef __attribute__((ext_vector_type(8)))  _Float16 v8h;
typedef __attribute__((ext_vector_type(8)))  float    v8f;

#define EMBED  1024
#define HEADS  16
#define HDIM   64
#define FDIM   128          // 2*HDIM hedgehog feature dim
#define NB     2
#define SEQ    2048
#define BQ     64           // q rows per workgroup (4 waves x 16)
#define BK     64           // k columns per iteration

// ---------------------------------------------------------------------------
// WMMA fragment helpers (CDNA5 ISA 7.12.2 layouts, wave32)
// ---------------------------------------------------------------------------

// A-matrix 16x32 f16, row-major source with leading dim `ld` (halves).
// lane 0-15 : row M=lane,    halves 0-7 = K 0-7,  halves 8-15 = K 16-23
// lane 16-31: row M=lane-16, halves 0-7 = K 8-15, halves 8-15 = K 24-31
__device__ __forceinline__ v16h load_a16x32(const _Float16* __restrict__ src, int ld) {
    const int lane = threadIdx.x & 31;
    const int m    = lane & 15;
    const int kb   = (lane & 16) ? 8 : 0;
    const _Float16* p = src + m * ld + kb;
    v8h lo = *(const v8h*)(p);
    v8h hi = *(const v8h*)(p + 16);
    v16h a;
#pragma unroll
    for (int i = 0; i < 8; ++i) { a[i] = lo[i]; a[8 + i] = hi[i]; }
    return a;
}

// B-matrix 32x16 f16 where the source is stored N-major (row = N, K contiguous)
// with leading dim `ld` (halves):  B[K][N] = src[N*ld + K].
// lane 0-15: N=lane, K 0-7,16-23; lane 16-31: N=lane-16, K 8-15,24-31.
__device__ __forceinline__ v16h load_b32x16(const _Float16* __restrict__ src, int ld) {
    const int lane = threadIdx.x & 31;
    const int n    = lane & 15;
    const int kb   = (lane & 16) ? 8 : 0;
    const _Float16* p = src + n * ld + kb;
    v8h lo = *(const v8h*)(p);
    v8h hi = *(const v8h*)(p + 16);
    v16h b;
#pragma unroll
    for (int i = 0; i < 8; ++i) { b[i] = lo[i]; b[8 + i] = hi[i]; }
    return b;
}

__device__ __forceinline__ v8f wmma_f16(v16h a, v16h b, v8f c) {
    return __builtin_amdgcn_wmma_f32_16x16x32_f16(false, a, false, b,
                                                  (short)0, c, false, false);
}

__device__ __forceinline__ v8f zero8() {
    v8f z;
#pragma unroll
    for (int i = 0; i < 8; ++i) z[i] = 0.0f;
    return z;
}

// Low 32 bits of a generic pointer into __shared__ = LDS byte address
// (shared aperture: high bits = SHARED_BASE, low 32 = wave-relative LDS offset).
__device__ __forceinline__ unsigned lds_addr(const void* p) {
    return (unsigned)(unsigned long long)p;
}

// ---------------------------------------------------------------------------
// Kernel 1: Wcomb[p] = Wf @ W[p]   (fold feature-map linear into projection)
// z = (x W^T) Wf^T = x (Wf W)^T
// grid (3,64), block 64
// ---------------------------------------------------------------------------
__global__ void hh_wcomb(const float* __restrict__ Wq, const float* __restrict__ Wk,
                         const float* __restrict__ Wv, const float* __restrict__ Wf,
                         float* __restrict__ Wcomb) {
    const int p = blockIdx.x;   // 0=q 1=k 2=v
    const int e = blockIdx.y;   // 0..63
    const int d = threadIdx.x;  // 0..63
    const float* W = (p == 0) ? Wq : (p == 1) ? Wk : Wv;
    float s = 0.0f;
#pragma unroll 8
    for (int t = 0; t < HDIM; ++t) s += Wf[e * HDIM + t] * W[t * HDIM + d];
    Wcomb[(p * HDIM + e) * HDIM + d] = s;
}

// ---------------------------------------------------------------------------
// Kernel 2: feature map.  F[n][h][l][0:64]=exp(z), [64:128]=exp(-z),
// z = x_head @ Wc^T + bf.  One 64-thread group per (n,l,h) row; block = 4 rows.
// grid NB*SEQ*HEADS/4, block 256
// ---------------------------------------------------------------------------
__global__ void hh_feature(const float* __restrict__ X, const float* __restrict__ Wc,
                           const float* __restrict__ bf, _Float16* __restrict__ F) {
    __shared__ float xs[256];
    const int t = threadIdx.x;
    xs[t] = X[(long)blockIdx.x * 256 + t];
    __syncthreads();
    const int rl = t >> 6;      // row within block 0..3
    const int e  = t & 63;
    const long g = (long)blockIdx.x * 4 + rl;   // flat (n,l,h)
    const int n   = (int)(g / (SEQ * HEADS));
    const int rem = (int)(g % (SEQ * HEADS));
    const int l   = rem / HEADS;
    const int h   = rem % HEADS;
    float z = bf[e];
    const float* w = Wc + e * HDIM;
    const float* x = xs + rl * HDIM;
#pragma unroll
    for (int d = 0; d < HDIM; ++d) z += x[d] * w[d];
    const long ob = (((long)(n * HEADS + h)) * SEQ + l) * FDIM;
    F[ob + e]        = (_Float16)__expf(z);
    F[ob + HDIM + e] = (_Float16)__expf(-z);
}

// ---------------------------------------------------------------------------
// Kernel 3: flash attention per (n,h).  4 waves, each owns 16 q-rows.
// K block staged with ASYNCcnt-tracked global_load_async_to_lds_b128,
// V block transposed manually (async can't transpose).
// grid (NB*HEADS, SEQ/BQ), block 128
// ---------------------------------------------------------------------------
__global__ void __launch_bounds__(128) hh_attn(const _Float16* __restrict__ Qf,
                                               const _Float16* __restrict__ Kf,
                                               const _Float16* __restrict__ Vf,
                                               _Float16* __restrict__ Oc) {
    __shared__ __align__(64) _Float16 Ks[BK][FDIM];      // K block, row-major [key][feat]
    __shared__ __align__(64) _Float16 Vt[FDIM][BK];      // V block transposed [feat][key]
    __shared__ __align__(64) _Float16 Pl[4][16][BK];     // per-wave P staging

    const int nh   = blockIdx.x;
    const int n    = nh / HEADS;
    const int h    = nh % HEADS;
    const int wave = threadIdx.x >> 5;
    const int lane = threadIdx.x & 31;
    const int q0   = blockIdx.y * BQ + wave * 16;

    // Q fragments for this wave's 16 rows: 4 chunks of K=32 features
    const _Float16* qbase = Qf + (((long)nh) * SEQ + q0) * FDIM;
    v16h aq[4];
#pragma unroll
    for (int c = 0; c < 4; ++c) aq[c] = load_a16x32(qbase + c * 32, FDIM);

    v8f   o[8];                       // 16 rows x 128 feat accumulator (8 n-tiles)
    float m[8], lsum[8];
#pragma unroll
    for (int ft = 0; ft < 8; ++ft) o[ft] = zero8();
#pragma unroll
    for (int r = 0; r < 8; ++r) { m[r] = -1e30f; lsum[r] = 0.0f; }

    const _Float16* kbase = Kf + ((long)nh) * SEQ * FDIM;
    const _Float16* vbase = Vf + ((long)nh) * SEQ * FDIM;
    const int rbase = (lane & 16) ? 8 : 0;   // C/D row offset for this lane
    const int nloc  = lane & 15;             // C/D column for this lane

    for (int k0 = 0; k0 < SEQ; k0 += BK) {
        __syncthreads();   // previous iteration's LDS reads complete
        {
            // 128 threads: each handles one key half-row of 64 features.
            const int key = threadIdx.x >> 1;
            const int fo  = (threadIdx.x & 1) * 64;
            const long gsrc = ((long)(k0 + key)) * FDIM + fo;

            // --- K: async global -> LDS copy (8 x 16B granules), ASYNCcnt ---
            const unsigned ldsK = lds_addr(&Ks[key][fo]);
            const unsigned voff = (unsigned)(gsrc * sizeof(_Float16));
#pragma unroll
            for (int c = 0; c < 8; ++c) {
                asm volatile("global_load_async_to_lds_b128 %0, %1, %2"
                             :: "v"(ldsK + c * 16), "v"(voff + c * 16), "s"(kbase)
                             : "memory");
            }

            // --- V: load rows, store transposed into LDS ---
            const _Float16* vp = vbase + gsrc;
#pragma unroll
            for (int c = 0; c < 8; ++c) {
                v8h vv = *(const v8h*)(vp + c * 8);
#pragma unroll
                for (int j = 0; j < 8; ++j) Vt[fo + c * 8 + j][key] = vv[j];
            }

            // --- prefetch next K/V block into cache while this one computes ---
            if (k0 + BK < SEQ) {
                __builtin_prefetch(kbase + gsrc + (long)BK * FDIM, 0, 1);
                __builtin_prefetch(vbase + gsrc + (long)BK * FDIM, 0, 1);
            }
        }
        asm volatile("s_wait_asynccnt 0x0" ::: "memory");
        __syncthreads();

        // S = Qf @ Kf^T for 4 n-tiles of 16 keys, K-dim = 128 feats
        v8f s[4];
#pragma unroll
        for (int tg = 0; tg < 4; ++tg) {
            v8f acc = zero8();
#pragma unroll
            for (int c = 0; c < 4; ++c) {
                v16h b = load_b32x16(&Ks[tg * 16][c * 32], FDIM);
                acc = wmma_f16(aq[c], b, acc);
            }
            s[tg] = acc;
        }

        // ---- online softmax (rows live across lanes of each 16-lane group) ----
        const float scale = 0.03125f;   // 1/sqrt(EMBED)
        float bm[8];
#pragma unroll
        for (int r = 0; r < 8; ++r) {
            float v = s[0][r] * scale;
#pragma unroll
            for (int tg = 1; tg < 4; ++tg) v = fmaxf(v, s[tg][r] * scale);
            v = fmaxf(v, __shfl_xor(v, 1, 32));
            v = fmaxf(v, __shfl_xor(v, 2, 32));
            v = fmaxf(v, __shfl_xor(v, 4, 32));
            v = fmaxf(v, __shfl_xor(v, 8, 32));
            bm[r] = v;
        }
        float al[8];
#pragma unroll
        for (int r = 0; r < 8; ++r) {
            float mn = fmaxf(m[r], bm[r]);
            al[r] = __expf(m[r] - mn);
            m[r]  = mn;
        }
        float p[4][8];
        float ps[8];
#pragma unroll
        for (int r = 0; r < 8; ++r) ps[r] = 0.0f;
#pragma unroll
        for (int tg = 0; tg < 4; ++tg)
#pragma unroll
            for (int r = 0; r < 8; ++r) {
                float pv = __expf(s[tg][r] * scale - m[r]);
                p[tg][r] = pv;
                ps[r] += pv;
            }
#pragma unroll
        for (int r = 0; r < 8; ++r) {
            float v = ps[r];
            v += __shfl_xor(v, 1, 32);
            v += __shfl_xor(v, 2, 32);
            v += __shfl_xor(v, 4, 32);
            v += __shfl_xor(v, 8, 32);
            lsum[r] = lsum[r] * al[r] + v;
        }
#pragma unroll
        for (int ft = 0; ft < 8; ++ft)
#pragma unroll
            for (int r = 0; r < 8; ++r) o[ft][r] *= al[r];

        // P: C/D layout -> A layout via per-wave LDS round trip
#pragma unroll
        for (int tg = 0; tg < 4; ++tg)
#pragma unroll
            for (int r = 0; r < 8; ++r)
                Pl[wave][rbase + r][tg * 16 + nloc] = (_Float16)p[tg][r];
        asm volatile("s_wait_dscnt 0x0" ::: "memory");

        // O += P @ V   (K-dim = 64 keys, 2 chunks of 32)
#pragma unroll
        for (int c = 0; c < 2; ++c) {
            v16h ap = load_a16x32(&Pl[wave][0][c * 32], BK);
#pragma unroll
            for (int ft = 0; ft < 8; ++ft) {
                v16h b = load_b32x16(&Vt[ft * 16][c * 32], BK);
                o[ft] = wmma_f16(ap, b, o[ft]);
            }
        }
    }

    // normalize and write Oc[n][l][h*128 + f] as f16 (A-operand of final GEMM)
    float inv[8];
#pragma unroll
    for (int r = 0; r < 8; ++r) inv[r] = 1.0f / lsum[r];
#pragma unroll
    for (int ft = 0; ft < 8; ++ft)
#pragma unroll
        for (int r = 0; r < 8; ++r) {
            const int q = q0 + rbase + r;
            const int f = ft * 16 + nloc;
            Oc[(((long)n * SEQ + q) * (HEADS * FDIM)) + h * FDIM + f] =
                (_Float16)(o[ft][r] * inv[r]);
        }
}

// ---------------------------------------------------------------------------
// Kernel 4: f32 -> f16 conversion (Wo)
// ---------------------------------------------------------------------------
__global__ void hh_cvt(const float* __restrict__ src, _Float16* __restrict__ dst, int n) {
    const int i = blockIdx.x * 256 + threadIdx.x;
    if (i < n) dst[i] = (_Float16)src[i];
}

// ---------------------------------------------------------------------------
// Kernel 5: out = Oc @ Wo^T + bo   (4096x1024x2048 GEMM, f16 in / f32 out)
// grid (NB*SEQ/64, EMBED/64), block 128 (4 waves x 16 rows, 64 cols each)
// ---------------------------------------------------------------------------
__global__ void __launch_bounds__(128) hh_out(const _Float16* __restrict__ A,
                                              const _Float16* __restrict__ B,
                                              const float* __restrict__ bo,
                                              float* __restrict__ out) {
    const int wave = threadIdx.x >> 5;
    const int lane = threadIdx.x & 31;
    const int row0 = blockIdx.x * 64 + wave * 16;
    const int col0 = blockIdx.y * 64;
    const int KD   = 2 * EMBED;   // 2048

    v8f acc[4];
#pragma unroll
    for (int t = 0; t < 4; ++t) acc[t] = zero8();

    for (int k = 0; k < KD; k += 32) {
        v16h a = load_a16x32(A + (long)row0 * KD + k, KD);
#pragma unroll
        for (int t = 0; t < 4; ++t) {
            v16h b = load_b32x16(B + (long)(col0 + t * 16) * KD + k, KD);
            acc[t] = wmma_f16(a, b, acc[t]);
        }
    }

    const int rbase = (lane & 16) ? 8 : 0;
    const int nloc  = lane & 15;
#pragma unroll
    for (int t = 0; t < 4; ++t)
#pragma unroll
        for (int r = 0; r < 8; ++r) {
            const int row = row0 + rbase + r;
            const int col = col0 + t * 16 + nloc;
            out[(long)row * EMBED + col] = acc[t][r] + bo[col];
        }
}

// ---------------------------------------------------------------------------
// Launch
// ---------------------------------------------------------------------------
extern "C" void kernel_launch(void* const* d_in, const int* in_sizes, int n_in,
                              void* d_out, int out_size, void* d_ws, size_t ws_size,
                              hipStream_t stream) {
    const float* values = (const float*)d_in[0];
    const float* keys   = (const float*)d_in[1];
    const float* query  = (const float*)d_in[2];
    const float* Wv     = (const float*)d_in[3];
    const float* Wk     = (const float*)d_in[4];
    const float* Wq     = (const float*)d_in[5];
    const float* Wf     = (const float*)d_in[6];
    const float* bf     = (const float*)d_in[7];
    const float* Wo     = (const float*)d_in[8];
    const float* bo     = (const float*)d_in[9];
    float* out = (float*)d_out;

    char*  ws  = (char*)d_ws;
    size_t off = 0;
    auto alloc = [&](size_t bytes) -> void* {
        void* p = ws + off;
        off += (bytes + 255) & ~(size_t)255;
        return p;
    };

    const size_t featBytes = (size_t)NB * HEADS * SEQ * FDIM * sizeof(_Float16); // 16 MB
    _Float16* Qf    = (_Float16*)alloc(featBytes);
    _Float16* Kf    = (_Float16*)alloc(featBytes);
    _Float16* Vf    = (_Float16*)alloc(featBytes);
    _Float16* Oc    = (_Float16*)alloc((size_t)NB * SEQ * (HEADS * FDIM) * sizeof(_Float16));
    _Float16* Wo16  = (_Float16*)alloc((size_t)EMBED * 2 * EMBED * sizeof(_Float16));
    float*    Wcomb = (float*)alloc(3 * HDIM * HDIM * sizeof(float));

    hh_wcomb<<<dim3(3, HDIM), HDIM, 0, stream>>>(Wq, Wk, Wv, Wf, Wcomb);

    const int featBlocks = NB * SEQ * HEADS / 4;   // 16384
    hh_feature<<<featBlocks, 256, 0, stream>>>(query,  Wcomb + 0 * HDIM * HDIM, bf, Qf);
    hh_feature<<<featBlocks, 256, 0, stream>>>(keys,   Wcomb + 1 * HDIM * HDIM, bf, Kf);
    hh_feature<<<featBlocks, 256, 0, stream>>>(values, Wcomb + 2 * HDIM * HDIM, bf, Vf);

    const int woElems = EMBED * 2 * EMBED;
    hh_cvt<<<(woElems + 255) / 256, 256, 0, stream>>>(Wo, Wo16, woElems);

    hh_attn<<<dim3(NB * HEADS, SEQ / BQ), 128, 0, stream>>>(Qf, Kf, Vf, Oc);

    hh_out<<<dim3(NB * SEQ / 64, EMBED / 64), 128, 0, stream>>>(Oc, Wo16, bo, out);
}
